// GGNNContext_45131516347042
// MI455X (gfx1250) — matrix operand order repaired and checked by reading, not compiled, not run on hardware
//
#include <hip/hip_runtime.h>
#include <hip/hip_bf16.h>
#include <math.h>

// ---------------------------------------------------------------------------
// Types for CDNA5 WMMA (gfx1250, wave32)
// ---------------------------------------------------------------------------
typedef __attribute__((ext_vector_type(16))) __bf16 v16bf;
typedef __attribute__((ext_vector_type(8)))  float  v8f;

union FragU { uint4 q[2]; v16bf v; };

#define TM 128
#define TN 128
#define TK 32
#define NTHREADS 512
#define CH ((TM * TK / 4) / NTHREADS)   // float4 chunks per thread per matrix (=2)

// Guarded float4 global load (cold path).
__device__ __forceinline__ float4 ld4g(const float* __restrict__ p, bool ok,
                                       int rem, bool al4) {
  float4 v = make_float4(0.f, 0.f, 0.f, 0.f);
  if (ok) {
    if (al4 && rem >= 4) {
      v = *(const float4*)p;
    } else {
      if (rem > 0) v.x = p[0];
      if (rem > 1) v.y = p[1];
      if (rem > 2) v.z = p[2];
      if (rem > 3) v.w = p[3];
    }
  }
  return v;
}

__device__ __forceinline__ void st4bf(__bf16* p, float4 v) {
  union { __bf16 h[4]; uint2 u; } pk;
  pk.h[0] = (__bf16)v.x; pk.h[1] = (__bf16)v.y;
  pk.h[2] = (__bf16)v.z; pk.h[3] = (__bf16)v.w;
  *(uint2*)p = pk.u;      // 8B LDS store
}

// ---------------------------------------------------------------------------
// Generic GEMM: C[M,N] = act( op(A)[M,K] @ op(B)[K,N] + bias (+C if addC) )
//   AT=0: A[m*lda+k]   AT=1: A[k*lda+m]
//   BT=0: B[k*ldb+n]   BT=1: B[n*ldb+k]
// fp32 global -> bf16 LDS staging -> v_wmma_f32_16x16x32_bf16, f32 accum.
// 16 waves/block; each wave owns 32x32 as 2x2 WMMA fragments per 32-K step.
// Block-uniform fast path: no guards in the steady-state loop.
// ---------------------------------------------------------------------------
template <int AT, int BT>
__global__ __launch_bounds__(NTHREADS)
void k_gemm_t(const float* __restrict__ A, int lda,
              const float* __restrict__ B, int ldb,
              float* __restrict__ C, int ldc,
              const float* __restrict__ bias,
              int M, int N, int K, int addC, int relu)
{
  __shared__ __align__(16) __bf16 sa[TM * TK];   // [m][k], k contiguous
  __shared__ __align__(16) __bf16 sb[TN * TK];   // [n][k], k contiguous

  const int bm = blockIdx.y * TM;
  const int bn = blockIdx.x * TN;
  const int tid  = threadIdx.x;
  const int wave = tid >> 5;
  const int lane = tid & 31;
  const int wr = wave >> 2;        // 0..3 -> 32 M-rows each
  const int wc = wave & 3;         // 0..3 -> 32 N-cols each
  const int lrow  = lane & 15;
  const int lhalf = lane >> 4;

  // ---- loop-invariant LDS addressing -----------------------------------
  int saOff[CH], sbOff[CH];
  #pragma unroll
  for (int i = 0; i < CH; ++i) {
    int c = tid + i * NTHREADS;
    saOff[i] = (AT == 0) ? ((c >> 3) * TK + (c & 7) * 4)
                         : (((c & 31) * 4) * TK + (c >> 5));
    sbOff[i] = (BT == 1) ? ((c >> 3) * TK + (c & 7) * 4)
                         : (((c & 31) * 4) * TK + (c >> 5));
  }
  const __bf16* paf[2];
  const __bf16* pbf[2];
  #pragma unroll
  for (int i = 0; i < 2; ++i) {
    paf[i] = &sa[(wr * 32 + i * 16 + lrow) * TK];
    pbf[i] = &sb[(wc * 32 + i * 16 + lrow) * TK];
  }

  v8f acc[2][2];
  #pragma unroll
  for (int i = 0; i < 2; ++i)
    #pragma unroll
    for (int j = 0; j < 2; ++j)
      #pragma unroll
      for (int r = 0; r < 8; ++r)
        acc[i][j][r] = 0.0f;

  // Fragment load + 4 WMMAs for the current LDS tiles.
  auto mma_step = [&]() {
    FragU fa[2], fb[2];
    #pragma unroll
    for (int i = 0; i < 2; ++i) {
      // A 16x32: lane(m=lrow): v0..3 K=lhalf*8+0..7, v4..7 K=16+lhalf*8+0..7
      fa[i].q[0] = *(const uint4*)(paf[i] + lhalf * 8);
      fa[i].q[1] = *(const uint4*)(paf[i] + 16 + lhalf * 8);
      // B 32x16: lane(n=lrow): v0..3 K=lhalf*8+0..7, v4..7 K=16+lhalf*8+0..7
      fb[i].q[0] = *(const uint4*)(pbf[i] + lhalf * 8);
      fb[i].q[1] = *(const uint4*)(pbf[i] + 16 + lhalf * 8);
    }
    #pragma unroll
    for (int i = 0; i < 2; ++i)
      #pragma unroll
      for (int j = 0; j < 2; ++j)
        acc[i][j] = __builtin_amdgcn_wmma_f32_16x16x32_bf16(
            false, fa[i].v, false, fb[j].v, (short)0, acc[i][j], false, false);
  };

  // Store a tile-chunk (registers -> bf16 LDS) for either layout.
  auto store_a = [&](int i, float4 v) {
    if (AT == 0) st4bf(&sa[saOff[i]], v);
    else {
      sa[saOff[i] + 0 * TK] = (__bf16)v.x;
      sa[saOff[i] + 1 * TK] = (__bf16)v.y;
      sa[saOff[i] + 2 * TK] = (__bf16)v.z;
      sa[saOff[i] + 3 * TK] = (__bf16)v.w;
    }
  };
  auto store_b = [&](int i, float4 v) {
    if (BT == 1) st4bf(&sb[sbOff[i]], v);
    else {
      sb[sbOff[i] + 0 * TK] = (__bf16)v.x;
      sb[sbOff[i] + 1 * TK] = (__bf16)v.y;
      sb[sbOff[i] + 2 * TK] = (__bf16)v.z;
      sb[sbOff[i] + 3 * TK] = (__bf16)v.w;
    }
  };

  const bool alA = ((lda & 3) == 0) && ((((uintptr_t)A) & 15) == 0);
  const bool alB = ((ldb & 3) == 0) && ((((uintptr_t)B) & 15) == 0);
  const bool fast = (bm + TM <= M) && (bn + TN <= N) &&
                    ((K & (TK - 1)) == 0) && alA && alB;

  if (fast) {
    // ---- FAST PATH: pointer-bumped unguarded b128 loads -----------------
    const float* pa[CH];
    const float* pb[CH];
    #pragma unroll
    for (int i = 0; i < CH; ++i) {
      int c = tid + i * NTHREADS;
      if (AT == 0) pa[i] = A + (unsigned)((unsigned)(bm + (c >> 3)) * lda + (c & 7) * 4);
      else         pa[i] = A + (unsigned)((unsigned)(c >> 5) * lda + bm + (c & 31) * 4);
      if (BT == 1) pb[i] = B + (unsigned)((unsigned)(bn + (c >> 3)) * ldb + (c & 7) * 4);
      else         pb[i] = B + (unsigned)((unsigned)(c >> 5) * ldb + bn + (c & 31) * 4);
    }
    const int aStep = (AT == 0) ? TK : TK * lda;
    const int bStep = (BT == 1) ? TK : TK * ldb;

    for (int k0 = 0; k0 < K; k0 += TK) {
      float4 va[CH], vb[CH];
      #pragma unroll
      for (int i = 0; i < CH; ++i) {
        va[i] = *(const float4*)pa[i];  pa[i] += aStep;
        vb[i] = *(const float4*)pb[i];  pb[i] += bStep;
      }
      #pragma unroll
      for (int i = 0; i < CH; ++i) { store_a(i, va[i]); store_b(i, vb[i]); }
      __syncthreads();
      mma_step();
      __syncthreads();
    }
  } else {
    // ---- SLOW PATH (edge tiles / odd leading dims) ----------------------
    for (int k0 = 0; k0 < K; k0 += TK) {
      float4 va[CH], vb[CH];
      #pragma unroll
      for (int i = 0; i < CH; ++i) {
        int c = tid + i * NTHREADS;
        if (AT == 0) {
          int row = c >> 3, kpos = (c & 7) * 4;
          int gm = bm + row, gk = k0 + kpos;
          va[i] = ld4g(A + (unsigned)((unsigned)gm * lda + gk), gm < M, K - gk, alA);
        } else {
          int p = (c & 31) * 4, kk = c >> 5;
          int gm0 = bm + p, gk = k0 + kk;
          va[i] = ld4g(A + (unsigned)((unsigned)gk * lda + gm0), gk < K, M - gm0, alA);
        }
        if (BT == 0) {
          int p = (c & 31) * 4, kk = c >> 5;
          int gn0 = bn + p, gk = k0 + kk;
          vb[i] = ld4g(B + (unsigned)((unsigned)gk * ldb + gn0), gk < K, N - gn0, alB);
        } else {
          int row = c >> 3, kpos = (c & 7) * 4;
          int gn = bn + row, gk = k0 + kpos;
          vb[i] = ld4g(B + (unsigned)((unsigned)gn * ldb + gk), gn < N, K - gk, alB);
        }
      }
      #pragma unroll
      for (int i = 0; i < CH; ++i) { store_a(i, va[i]); store_b(i, vb[i]); }
      __syncthreads();
      mma_step();
      __syncthreads();
    }
  }

  // ---- epilogue: C/D layout: VGPR r -> M = lhalf*8 + r, N = lrow ----------
  if (bm + TM <= M && bn + TN <= N) {
    #pragma unroll
    for (int i = 0; i < 2; ++i) {
      #pragma unroll
      for (int j = 0; j < 2; ++j) {
        int gn = bn + wc * 32 + j * 16 + lrow;
        float bb = bias ? bias[gn] : 0.0f;
        #pragma unroll
        for (int r = 0; r < 8; ++r) {
          int gm = bm + wr * 32 + i * 16 + lhalf * 8 + r;
          float v = acc[i][j][r] + bb;
          if (addC) v += C[(size_t)gm * ldc + gn];
          if (relu && v < 0.0f) v = 0.0f;
          C[(size_t)gm * ldc + gn] = v;
        }
      }
    }
  } else {
    #pragma unroll
    for (int i = 0; i < 2; ++i) {
      #pragma unroll
      for (int j = 0; j < 2; ++j) {
        int gn = bn + wc * 32 + j * 16 + lrow;
        #pragma unroll
        for (int r = 0; r < 8; ++r) {
          int gm = bm + wr * 32 + i * 16 + lhalf * 8 + r;
          if (gm < M && gn < N) {
            float v = acc[i][j][r];
            if (bias) v += bias[gn];
            if (addC) v += C[(size_t)gm * ldc + gn];
            if (relu && v < 0.0f) v = 0.0f;
            C[(size_t)gm * ldc + gn] = v;
          }
        }
      }
    }
  }
}

// ---------------------------------------------------------------------------
// Elementwise / helper kernels
// ---------------------------------------------------------------------------
__global__ void k_fill(float* p, size_t n, float v) {
  size_t i = (size_t)blockIdx.x * blockDim.x + threadIdx.x;
  if (i < n) p[i] = v;
}

__global__ void k_onehot(float* A, const int* idx, int which, int ncols, int R) {
  int r = blockIdx.x * blockDim.x + threadIdx.x;
  if (r < R) A[(size_t)r * ncols + idx[r * 2 + which]] = 1.0f;
}

__global__ void k_gather(float* dst, int ldd, const float* src, int lds_,
                         const int* idx, int which, int Dd, int R) {
  size_t i = (size_t)blockIdx.x * blockDim.x + threadIdx.x;
  if (i < (size_t)R * Dd) {
    int r = (int)(i / Dd), c = (int)(i % Dd);
    dst[(size_t)r * ldd + c] = src[(size_t)idx[r * 2 + which] * lds_ + c];
  }
}

__global__ void k_gru_zr(float* g0, const float* g1, const float* node,
                         float* rn, size_t n) {
  size_t i = (size_t)blockIdx.x * blockDim.x + threadIdx.x;
  if (i < n) {
    float z = 1.0f / (1.0f + __expf(-g0[i]));
    g0[i] = z;
    float r = 1.0f / (1.0f + __expf(-g1[i]));
    rn[i] = r * node[i];
  }
}

__global__ void k_gru_fin(float* node, const float* z, const float* hpre, size_t n) {
  size_t i = (size_t)blockIdx.x * blockDim.x + threadIdx.x;
  if (i < n) {
    float h = tanhf(hpre[i]);
    float zz = z[i];
    node[i] = (1.0f - zz) * node[i] + zz * h;
  }
}

__global__ void k_softmax(const float* __restrict__ in, float* __restrict__ out, int cols) {
  int row = blockIdx.x;
  const float* x = in + (size_t)row * cols;
  float* y = out + (size_t)row * cols;
  __shared__ float red[128];
  int t = threadIdx.x;
  float mx = -INFINITY;
  for (int c = t; c < cols; c += 128) mx = fmaxf(mx, x[c]);
  red[t] = mx; __syncthreads();
  for (int s = 64; s > 0; s >>= 1) { if (t < s) red[t] = fmaxf(red[t], red[t + s]); __syncthreads(); }
  mx = red[0]; __syncthreads();
  float sum = 0.0f;
  for (int c = t; c < cols; c += 128) sum += __expf(x[c] - mx);
  red[t] = sum; __syncthreads();
  for (int s = 64; s > 0; s >>= 1) { if (t < s) red[t] += red[t + s]; __syncthreads(); }
  sum = red[0];
  float inv = 1.0f / sum;
  for (int c = t; c < cols; c += 128) y[c] = __expf(x[c] - mx) * inv;
}

// ---------------------------------------------------------------------------
// Host-side GEMM dispatch
// ---------------------------------------------------------------------------
static inline void gemm_disp(hipStream_t s,
                             const float* A, int lda, int aT,
                             const float* B, int ldb, int bT,
                             float* C, int ldc, const float* bias,
                             int M, int N, int K, int addC, int relu) {
  dim3 g((N + TN - 1) / TN, (M + TM - 1) / TM);
  dim3 b(NTHREADS);
  if (!aT && !bT)
    k_gemm_t<0, 0><<<g, b, 0, s>>>(A, lda, B, ldb, C, ldc, bias, M, N, K, addC, relu);
  else if (aT && !bT)
    k_gemm_t<1, 0><<<g, b, 0, s>>>(A, lda, B, ldb, C, ldc, bias, M, N, K, addC, relu);
  else if (!aT && bT)
    k_gemm_t<0, 1><<<g, b, 0, s>>>(A, lda, B, ldb, C, ldc, bias, M, N, K, addC, relu);
  else
    k_gemm_t<1, 1><<<g, b, 0, s>>>(A, lda, B, ldb, C, ldc, bias, M, N, K, addC, relu);
}

// ---------------------------------------------------------------------------
// Host orchestration
// ---------------------------------------------------------------------------
extern "C" void kernel_launch(void* const* d_in, const int* in_sizes, int n_in,
                              void* d_out, int out_size, void* d_ws, size_t ws_size,
                              hipStream_t stream) {
  (void)in_sizes; (void)n_in; (void)out_size; (void)ws_size;
  const int Nn = 512, R = 4096, D = 1024, D2 = 512, D4 = 256;
  const int CE = 151, CR = 51, MO = 1792, MI = 768, T = 3;

  const float* roi        = (const float*)d_in[0];
  const float* uni        = (const float*)d_in[1];
  const int*   rel        = (const int*)  d_in[2];
  const float* obj_logits = (const float*)d_in[3];
  const float* emb_ent    = (const float*)d_in[4];
  const float* emb_pred   = (const float*)d_in[5];
  const float* adj_e2e    = (const float*)d_in[6];
  const float* adj_e2p    = (const float*)d_in[7];
  const float* adj_p2e    = (const float*)d_in[8];
  const float* adj_p2p    = (const float*)d_in[9];
  const float* init_ent_w = (const float*)d_in[10];
  const float* init_ent_b = (const float*)d_in[11];
  const float* init_pred_w= (const float*)d_in[12];
  const float* init_pred_b= (const float*)d_in[13];
  const float* send_w1    = (const float*)d_in[14];
  const float* send_b1    = (const float*)d_in[15];
  const float* send_w2    = (const float*)d_in[16];
  const float* send_b2    = (const float*)d_in[17];
  const float* rw1[4] = {(const float*)d_in[18], (const float*)d_in[22], (const float*)d_in[26], (const float*)d_in[30]};
  const float* rb1[4] = {(const float*)d_in[19], (const float*)d_in[23], (const float*)d_in[27], (const float*)d_in[31]};
  const float* rw2[4] = {(const float*)d_in[20], (const float*)d_in[24], (const float*)d_in[28], (const float*)d_in[32]};
  const float* rb2[4] = {(const float*)d_in[21], (const float*)d_in[25], (const float*)d_in[29], (const float*)d_in[33]};
  const float* gW  = (const float*)d_in[34];
  const float* gU  = (const float*)d_in[35];
  const float* gbW = (const float*)d_in[36];
  const float* gbU = (const float*)d_in[37];
  const float* ow1 = (const float*)d_in[38];
  const float* ob1 = (const float*)d_in[39];
  const float* ow2 = (const float*)d_in[40];
  const float* ob2 = (const float*)d_in[41];
  float* out = (float*)d_out;                // [512*151] ++ [4096*51]

  // ---- workspace bump allocator (floats) --------------------------------
  float* wsp = (float*)d_ws;
  size_t off = 0;
  auto alloc = [&](size_t n) { float* p = wsp + off; off += n; return p; };

  float* oe  = alloc((size_t)CE * D);
  float* op_ = alloc((size_t)CR * D);
  float* ie_ = alloc((size_t)Nn * D);
  float* ip_ = alloc((size_t)R  * D);
  float* m_oe = alloc((size_t)CE * D4);
  float* m_op = alloc((size_t)CR * D4);
  float* m_ie = alloc((size_t)Nn * D4);
  float* m_ip = alloc((size_t)R  * D4);
  float* rcv_oe = alloc((size_t)CE * MO);
  float* rcv_op = alloc((size_t)CR * MO);
  float* rcv_ie = alloc((size_t)Nn * MI);
  float* rcv_ip = alloc((size_t)R  * MI);
  float* hid = alloc((size_t)R * D);         // biggest hidden: [4096,1024]
  float* msg = alloc((size_t)R * D);         // also reused as rn in GRU
  float* g0  = alloc((size_t)R * D);
  float* g1  = alloc((size_t)R * D);
  float* g2  = alloc((size_t)R * D);
  float* As  = alloc((size_t)R * Nn);
  float* Ao  = alloc((size_t)R * Nn);
  float* ieo = alloc((size_t)Nn * D);
  float* oeo = alloc((size_t)CE * D);
  float* ipo = alloc((size_t)R  * D);
  float* opo = alloc((size_t)CR * D);
  float* Eent  = alloc((size_t)Nn * CE);
  float* Epred = alloc((size_t)R  * CR);

  auto gemm = [&](const float* A, int lda, int aT, const float* B, int ldb, int bT,
                  float* C, int ldc, const float* bias, int M, int N, int K,
                  int addC, int relu) {
    gemm_disp(stream, A, lda, aT, B, ldb, bT, C, ldc, bias, M, N, K, addC, relu);
  };
  auto fill = [&](float* p, size_t n, float v) {
    k_fill<<<(unsigned)((n + 255) / 256), 256, 0, stream>>>(p, n, v);
  };
  auto ew = [&](size_t n) { return (unsigned)((n + 255) / 256); };

  // ---- initialization ---------------------------------------------------
  hipMemcpyAsync(ie_, roi, (size_t)Nn * D * sizeof(float), hipMemcpyDeviceToDevice, stream);
  hipMemcpyAsync(ip_, uni, (size_t)R  * D * sizeof(float), hipMemcpyDeviceToDevice, stream);
  gemm(emb_ent, 300, 0, init_ent_w, D, 0, oe,  D, init_ent_b,  CE, D, 300, 0, 0);
  gemm(emb_pred,300, 0, init_pred_w,D, 0, op_, D, init_pred_b, CR, D, 300, 0, 0);
  k_softmax<<<Nn, 128, 0, stream>>>(obj_logits, Eent, CE);
  fill(Epred, (size_t)R * CR, 0.0f);
  fill(As, (size_t)R * Nn, 0.0f);
  fill(Ao, (size_t)R * Nn, 0.0f);
  k_onehot<<<(R + 255) / 256, 256, 0, stream>>>(As, rel, 0, Nn, R);
  k_onehot<<<(R + 255) / 256, 256, 0, stream>>>(Ao, rel, 1, Nn, R);

  struct NodeCfg { float* state; float* m; float* rcv; int n; int Mw; };
  NodeCfg nodes[4] = {
    { oe,  m_oe, rcv_oe, CE, MO },
    { op_, m_op, rcv_op, CR, MO },
    { ie_, m_ie, rcv_ie, Nn, MI },
    { ip_, m_ip, rcv_ip, R,  MI },
  };

  for (int step = 0; step < T; ++step) {
    // ---- send MLPs (d -> d2 -> d4, relu both) ---------------------------
    for (int t = 0; t < 4; ++t) {
      const float* w1 = send_w1 + (size_t)t * D * D2;
      const float* b1 = send_b1 + (size_t)t * D2;
      const float* w2 = send_w2 + (size_t)t * D2 * D4;
      const float* b2 = send_b2 + (size_t)t * D4;
      gemm(nodes[t].state, D, 0, w1, D2, 0, hid, D2, b1, nodes[t].n, D2, D, 0, 1);
      gemm(hid, D2, 0, w2, D4, 0, nodes[t].m, D4, b2, nodes[t].n, D4, D2, 0, 1);
    }

    // ---- receive aggregation --------------------------------------------
    for (int k = 0; k < 3; ++k) {  // rcv_oe: e2e^T, p2e^T
      gemm(adj_e2e + (size_t)k * CE * CE, CE, 1, m_oe, D4, 0,
           rcv_oe + (size_t)k * D4, MO, nullptr, CE, D4, CE, 0, 0);
      gemm(adj_p2e + (size_t)k * CR * CE, CE, 1, m_op, D4, 0,
           rcv_oe + 768 + (size_t)k * D4, MO, nullptr, CE, D4, CR, 0, 0);
    }
    gemm(Eent, CE, 1, m_ie, D4, 0, rcv_oe + 1536, MO, nullptr, CE, D4, Nn, 0, 0);

    for (int k = 0; k < 3; ++k) {  // rcv_op: e2p^T, p2p^T
      gemm(adj_e2p + (size_t)k * CE * CR, CR, 1, m_oe, D4, 0,
           rcv_op + (size_t)k * D4, MO, nullptr, CR, D4, CE, 0, 0);
      gemm(adj_p2p + (size_t)k * CR * CR, CR, 1, m_op, D4, 0,
           rcv_op + 768 + (size_t)k * D4, MO, nullptr, CR, D4, CR, 0, 0);
    }
    gemm(Epred, CR, 1, m_ip, D4, 0, rcv_op + 1536, MO, nullptr, CR, D4, R, 0, 0);

    // rcv_ie: A_s^T @ m_ip | A_o^T @ m_ip | E_ent @ m_oe
    gemm(As, Nn, 1, m_ip, D4, 0, rcv_ie,        MI, nullptr, Nn, D4, R,  0, 0);
    gemm(Ao, Nn, 1, m_ip, D4, 0, rcv_ie + 256,  MI, nullptr, Nn, D4, R,  0, 0);
    gemm(Eent, CE, 0, m_oe, D4, 0, rcv_ie + 512, MI, nullptr, Nn, D4, CE, 0, 0);

    // rcv_ip: gather(m_ie, subj) | gather(m_ie, obj) | E_pred @ m_op
    k_gather<<<ew((size_t)R * D4), 256, 0, stream>>>(rcv_ip,        MI, m_ie, D4, rel, 0, D4, R);
    k_gather<<<ew((size_t)R * D4), 256, 0, stream>>>(rcv_ip + 256,  MI, m_ie, D4, rel, 1, D4, R);
    gemm(Epred, CR, 0, m_op, D4, 0, rcv_ip + 512, MI, nullptr, R, D4, CR, 0, 0);

    // ---- per-type receive MLP + GRU update ------------------------------
    for (int t = 0; t < 4; ++t) {
      int n = nodes[t].n, Mw = nodes[t].Mw;
      float* state = nodes[t].state;
      // msg = relu(relu(rcv@w1+b1)@w2+b2)
      gemm(nodes[t].rcv, Mw, 0, rw1[t], Mw, 0, hid, Mw, rb1[t], n, Mw, Mw, 0, 1);
      gemm(hid, Mw, 0, rw2[t], D, 0, msg, D, rb2[t], n, D, Mw, 0, 1);
      // GRU gates
      const float* W0 = gW + (size_t)(t * 3 + 0) * D * D;
      const float* W1 = gW + (size_t)(t * 3 + 1) * D * D;
      const float* W2 = gW + (size_t)(t * 3 + 2) * D * D;
      const float* U0 = gU + (size_t)(t * 3 + 0) * D * D;
      const float* U1 = gU + (size_t)(t * 3 + 1) * D * D;
      const float* U2 = gU + (size_t)(t * 3 + 2) * D * D;
      gemm(msg,   D, 0, W0, D, 0, g0, D, gbW + (size_t)(t * 3 + 0) * D, n, D, D, 0, 0);
      gemm(state, D, 0, U0, D, 0, g0, D, gbU + (size_t)(t * 3 + 0) * D, n, D, D, 1, 0);
      gemm(msg,   D, 0, W1, D, 0, g1, D, gbW + (size_t)(t * 3 + 1) * D, n, D, D, 0, 0);
      gemm(state, D, 0, U1, D, 0, g1, D, gbU + (size_t)(t * 3 + 1) * D, n, D, D, 1, 0);
      gemm(msg,   D, 0, W2, D, 0, g2, D, gbW + (size_t)(t * 3 + 2) * D, n, D, D, 0, 0);
      // z in g0, rn = sigmoid(g1)*state reuses msg buffer
      k_gru_zr<<<ew((size_t)n * D), 256, 0, stream>>>(g0, g1, state, msg, (size_t)n * D);
      gemm(msg, D, 0, U2, D, 0, g2, D, gbU + (size_t)(t * 3 + 2) * D, n, D, D, 1, 0);
      k_gru_fin<<<ew((size_t)n * D), 256, 0, stream>>>(state, g0, g2, (size_t)n * D);
    }

    // ---- output projections + logits ------------------------------------
    struct OutCfg { const float* x; float* y; int n; int w; };
    OutCfg outs[4] = {
      { ie_, ieo, Nn, 0 }, { oe, oeo, CE, 1 }, { ip_, ipo, R, 2 }, { op_, opo, CR, 3 },
    };
    for (int t = 0; t < 4; ++t) {
      const float* w1 = ow1 + (size_t)outs[t].w * D * D;
      const float* b1 = ob1 + (size_t)outs[t].w * D;
      const float* w2 = ow2 + (size_t)outs[t].w * D * D;
      const float* b2 = ob2 + (size_t)outs[t].w * D;
      gemm(outs[t].x, D, 0, w1, D, 0, hid, D, b1, outs[t].n, D, D, 0, 1);
      gemm(hid, D, 0, w2, D, 0, outs[t].y, D, b2, outs[t].n, D, D, 0, 0);
    }
    // ent_logits = ieo @ oeo^T ; pred_logits = ipo @ opo^T (into d_out)
    gemm(ieo, D, 0, oeo, D, 1, out,                     CE, nullptr, Nn, CE, D, 0, 0);
    gemm(ipo, D, 0, opo, D, 1, out + (size_t)Nn * CE,   CR, nullptr, R,  CR, D, 0, 0);

    // ---- refresh bridge matrices ----------------------------------------
    k_softmax<<<Nn, 128, 0, stream>>>(out,                   Eent,  CE);
    k_softmax<<<R,  128, 0, stream>>>(out + (size_t)Nn * CE, Epred, CR);
  }
}
